// SpatialAttention_45157286150596
// MI455X (gfx1250) — matrix operand order repaired and checked by reading, not compiled, not run on hardware
//
#include <hip/hip_runtime.h>
#include <hip/hip_bf16.h>

typedef __attribute__((ext_vector_type(16))) __bf16 v16bf;
typedef __attribute__((ext_vector_type(4)))  __bf16 v4bf;
typedef __attribute__((ext_vector_type(2)))  __bf16 v2bf;
typedef __attribute__((ext_vector_type(8)))  float  v8f;

#define LN_EPS 1e-5f
#define BTILE (128 * 32)   // bf16 elements per B tile

__device__ __forceinline__ v16bf pack_a(float4 a0, float4 a1, float4 a2, float4 a3) {
  v16bf r;
  r[0]=(__bf16)a0.x;  r[1]=(__bf16)a0.y;  r[2]=(__bf16)a0.z;  r[3]=(__bf16)a0.w;
  r[4]=(__bf16)a1.x;  r[5]=(__bf16)a1.y;  r[6]=(__bf16)a1.z;  r[7]=(__bf16)a1.w;
  r[8]=(__bf16)a2.x;  r[9]=(__bf16)a2.y;  r[10]=(__bf16)a2.z; r[11]=(__bf16)a2.w;
  r[12]=(__bf16)a3.x; r[13]=(__bf16)a3.y; r[14]=(__bf16)a3.z; r[15]=(__bf16)a3.w;
  return r;
}

// Stage 32(K) x 128(N) tile from bf16 B stored [K,N] row-major into LDS [n][k]
// (transposed). 256 threads, 8B loads, 4B interleaved stores.
__device__ __forceinline__ void stage_kn(__bf16* __restrict__ bs,
                                         const __bf16* __restrict__ B,
                                         int ldb, int kc, int n0, int t) {
#pragma unroll
  for (int rep = 0; rep < 2; ++rep) {
    const int kp2 = rep * 8 + (t >> 5);        // k-pair index 0..15
    const int n   = (t & 31) << 2;
    const __bf16* src = B + (long)(kc + 2*kp2) * ldb + n0 + n;
    const v4bf r0 = *(const v4bf*)(src);
    const v4bf r1 = *(const v4bf*)(src + ldb);
    v2bf p;
    p[0]=r0[0]; p[1]=r1[0]; *(v2bf*)&bs[(n+0)*32 + 2*kp2] = p;
    p[0]=r0[1]; p[1]=r1[1]; *(v2bf*)&bs[(n+1)*32 + 2*kp2] = p;
    p[0]=r0[2]; p[1]=r1[2]; *(v2bf*)&bs[(n+2)*32 + 2*kp2] = p;
    p[0]=r0[3]; p[1]=r1[3]; *(v2bf*)&bs[(n+3)*32 + 2*kp2] = p;
  }
}

// Stage from bf16 B stored [N,K] row-major (multiply by B^T): straight copy.
__device__ __forceinline__ void stage_nk(__bf16* __restrict__ bs,
                                         const __bf16* __restrict__ B,
                                         int ldb, int kc, int n0, int t) {
#pragma unroll
  for (int rep = 0; rep < 4; ++rep) {
    const int n  = rep * 32 + (t >> 3);
    const int kq = (t & 7) << 2;
    *(v4bf*)&bs[n * 32 + kq] = *(const v4bf*)(B + (long)(n0 + n) * ldb + kc + kq);
  }
}

// ---------------------------------------------------------------------------
// Generic batched GEMM: C = alpha * A(f32) @ B(bf16), bf16 WMMA, f32 accum.
// 256 thr = 8 waves; tile 128(M) x 128(N); K step 32; double-buffered B tile.
// out_bf16: write C as bf16 (row-major, ldc) instead of f32.
// ---------------------------------------------------------------------------
__global__ __launch_bounds__(256)
void gemm_bf16_kernel(const float* __restrict__ A, const __bf16* __restrict__ B,
                      void* __restrict__ Cv,
                      int lda, int ldb, int ldc,
                      long sA, long sB, long sC,
                      int K, float alpha, int b_is_nk, int out_bf16)
{
  __shared__ __align__(32) __bf16 Bs[2 * BTILE];   // 16 KB
  const int n0 = blockIdx.x * 128;
  const int m0 = blockIdx.y * 128;
  A += (long)blockIdx.z * sA;
  B += (long)blockIdx.z * sB;

  const int t    = threadIdx.x;
  const int lane = t & 31;
  const int wv   = t >> 5;
  const int mrow = m0 + wv * 16 + (lane & 15);
  const int kb   = (lane >> 4) << 4;          // 0 or 16 (A-frag K sub-chunk)
  const int bofs = (lane & 15) * 32 + ((lane >> 4) << 4); // B-frag base in tile

  v8f acc[8];
  const v8f vzero = {0,0,0,0,0,0,0,0};
#pragma unroll
  for (int i = 0; i < 8; ++i) acc[i] = vzero;

  const int nsteps = K >> 5;
  if (b_is_nk) stage_nk(Bs, B, ldb, 0, n0, t);
  else         stage_kn(Bs, B, ldb, 0, n0, t);

  const float* arow = A + (long)mrow * lda + kb;

  for (int s = 0; s < nsteps; ++s) {
    __syncthreads();
    // stage next K-step into the other buffer (global loads overlap compute)
    if (s + 1 < nsteps) {
      __bf16* nb = Bs + ((s + 1) & 1) * BTILE;
      if (b_is_nk) stage_nk(nb, B, ldb, (s + 1) << 5, n0, t);
      else         stage_kn(nb, B, ldb, (s + 1) << 5, n0, t);
    }
    // prefetch A two K-steps ahead (global_prefetch_b8)
    if (s + 2 < nsteps) __builtin_prefetch((const void*)(arow + ((s + 2) << 5)), 0, 0);

    const __bf16* bsrc = Bs + (s & 1) * BTILE;
    const float* ap = arow + (s << 5);
    float4 a0 = *(const float4*)(ap);
    float4 a1 = *(const float4*)(ap + 4);
    float4 a2 = *(const float4*)(ap + 8);
    float4 a3 = *(const float4*)(ap + 12);
    v16bf af = pack_a(a0, a1, a2, a3);

    v16bf bfr[8];
#pragma unroll
    for (int nt = 0; nt < 8; ++nt)
      bfr[nt] = *(const v16bf*)(bsrc + nt * 16 * 32 + bofs);
#pragma unroll
    for (int nt = 0; nt < 8; ++nt)
      acc[nt] = __builtin_amdgcn_wmma_f32_16x16x32_bf16(
          false, af, false, bfr[nt], (short)0, acc[nt], false, false);
  }

  const int rbase = m0 + wv * 16 + ((lane >> 4) << 3);
  if (out_bf16) {
    __bf16* Cb = (__bf16*)Cv + (long)blockIdx.z * sC;
#pragma unroll
    for (int nt = 0; nt < 8; ++nt) {
      int col = n0 + nt * 16 + (lane & 15);
#pragma unroll
      for (int r = 0; r < 8; ++r)
        Cb[(long)(rbase + r) * ldc + col] = (__bf16)(alpha * acc[nt][r]);
    }
  } else {
    float* Cf = (float*)Cv + (long)blockIdx.z * sC;
#pragma unroll
    for (int nt = 0; nt < 8; ++nt) {
      int col = n0 + nt * 16 + (lane & 15);
#pragma unroll
      for (int r = 0; r < 8; ++r)
        Cf[(long)(rbase + r) * ldc + col] = alpha * acc[nt][r];
    }
  }
}

// ---------------------------------------------------------------------------
// 3x3 SAME conv as implicit GEMM. IN: [8,32,32,512] f32, CW: bf16 [9][512][512].
// Flattened (tap, kc) loop of 144 steps, double-buffered weight tile.
// ---------------------------------------------------------------------------
__global__ __launch_bounds__(256)
void conv3x3_kernel(const float* __restrict__ IN, const __bf16* __restrict__ CW,
                    const float* __restrict__ CB, float* __restrict__ OUT)
{
  __shared__ __align__(32) __bf16 Bs[2 * BTILE];
  const int n0 = blockIdx.x * 128;
  const int m0 = blockIdx.y * 128;
  const int t    = threadIdx.x;
  const int lane = t & 31;
  const int wv   = t >> 5;
  const int mrow = m0 + wv * 16 + (lane & 15);
  const int kb   = (lane >> 4) << 4;
  const int bofs = (lane & 15) * 32 + ((lane >> 4) << 4);
  const int bidx = mrow >> 10, pos = mrow & 1023, yy = pos >> 5, xx = pos & 31;

  v8f acc[8];
  const v8f vzero = {0,0,0,0,0,0,0,0};
#pragma unroll
  for (int i = 0; i < 8; ++i) acc[i] = vzero;

  const int nsteps = 9 * 16;   // 9 taps x (512/32) K-chunks
  stage_kn(Bs, CW, 512, 0, n0, t);   // step 0: tap 0, kc 0

  for (int s = 0; s < nsteps; ++s) {
    __syncthreads();
    if (s + 1 < nsteps) {
      const int tap1 = (s + 1) >> 4;
      const int kc1  = ((s + 1) & 15) << 5;
      stage_kn(Bs + ((s + 1) & 1) * BTILE, CW + (long)tap1 * 512 * 512, 512, kc1, n0, t);
    }

    const int tap = s >> 4;
    const int kc  = (s & 15) << 5;
    const int dy = tap / 3 - 1, dx = tap % 3 - 1;
    const int y2 = yy + dy, x2 = xx + dx;
    const bool valid = ((unsigned)y2 < 32u) && ((unsigned)x2 < 32u);
    const float* ap = IN + ((long)bidx * 1024 + y2 * 32 + x2) * 512 + kc + kb;
    // prefetch the same spatial position two K-chunks ahead within this tap
    if (valid && (s & 15) < 14) __builtin_prefetch((const void*)(ap + 64), 0, 0);

    const float4 z = make_float4(0.f, 0.f, 0.f, 0.f);
    float4 a0 = valid ? *(const float4*)(ap)      : z;
    float4 a1 = valid ? *(const float4*)(ap + 4)  : z;
    float4 a2 = valid ? *(const float4*)(ap + 8)  : z;
    float4 a3 = valid ? *(const float4*)(ap + 12) : z;
    v16bf af = pack_a(a0, a1, a2, a3);

    const __bf16* bsrc = Bs + (s & 1) * BTILE;
    v16bf bfr[8];
#pragma unroll
    for (int nt = 0; nt < 8; ++nt)
      bfr[nt] = *(const v16bf*)(bsrc + nt * 16 * 32 + bofs);
#pragma unroll
    for (int nt = 0; nt < 8; ++nt)
      acc[nt] = __builtin_amdgcn_wmma_f32_16x16x32_bf16(
          false, af, false, bfr[nt], (short)0, acc[nt], false, false);
  }

  const int rbase = m0 + wv * 16 + ((lane >> 4) << 3);
#pragma unroll
  for (int nt = 0; nt < 8; ++nt) {
    int col = n0 + nt * 16 + (lane & 15);
    float bias = CB[col];
#pragma unroll
    for (int r = 0; r < 8; ++r) {
      OUT[(long)(rbase + r) * 512 + col] = acc[nt][r] + bias;
    }
  }
}

// ---------------------------------------------------------------------------
// f32 -> bf16 cast (n must be a multiple of 4)
// ---------------------------------------------------------------------------
__global__ __launch_bounds__(256)
void cast_bf16_kernel(const float* __restrict__ in, __bf16* __restrict__ out, long n)
{
  long i = ((long)blockIdx.x * 256 + threadIdx.x) * 4;
  if (i < n) {
    float4 v = *(const float4*)(in + i);
    v4bf p;
    p[0]=(__bf16)v.x; p[1]=(__bf16)v.y; p[2]=(__bf16)v.z; p[3]=(__bf16)v.w;
    *(v4bf*)(out + i) = p;
  }
}

// ---------------------------------------------------------------------------
// Row softmax over 1024 columns, in place. One block (256 thr) per row.
// ---------------------------------------------------------------------------
__global__ __launch_bounds__(256)
void softmax_kernel(float* __restrict__ scores)
{
  __shared__ float red[256];
  float* p = scores + (long)blockIdx.x * 1024;
  const int t = threadIdx.x;

  float v[4], mx = -3.4e38f;
#pragma unroll
  for (int i = 0; i < 4; ++i) { v[i] = p[t + i * 256]; mx = fmaxf(mx, v[i]); }
  red[t] = mx; __syncthreads();
  for (int s = 128; s > 0; s >>= 1) { if (t < s) red[t] = fmaxf(red[t], red[t+s]); __syncthreads(); }
  mx = red[0]; __syncthreads();

  float sum = 0.f;
#pragma unroll
  for (int i = 0; i < 4; ++i) { v[i] = __expf(v[i] - mx); sum += v[i]; }
  red[t] = sum; __syncthreads();
  for (int s = 128; s > 0; s >>= 1) { if (t < s) red[t] += red[t+s]; __syncthreads(); }
  const float inv = 1.0f / red[0];
#pragma unroll
  for (int i = 0; i < 4; ++i) p[t + i * 256] = v[i] * inv;
}

// ---------------------------------------------------------------------------
__global__ __launch_bounds__(512)
void se_pool_kernel(const float* __restrict__ co, float* __restrict__ pooled)
{
  const int b = blockIdx.x, d = threadIdx.x;
  const float* p = co + (long)b * 1024 * 512 + d;
  float s = 0.f;
  for (int n = 0; n < 1024; ++n) s += p[n * 512];
  pooled[b * 512 + d] = s * (1.0f / 1024.0f);
}

// ---------------------------------------------------------------------------
__global__ __launch_bounds__(128)
void se_mlp_kernel(const float* __restrict__ pooled,
                   const float* __restrict__ w1, const float* __restrict__ b1,
                   const float* __restrict__ w2, const float* __restrict__ b2,
                   float* __restrict__ gate)
{
  __shared__ float h[128];
  const int b = blockIdx.x, j = threadIdx.x;
  const float* pb = pooled + b * 512;

  float s = b1[j];
  for (int d = 0; d < 512; ++d) s += pb[d] * w1[d * 128 + j];
  h[j] = s / (1.0f + __expf(-s));            // silu
  __syncthreads();

  for (int d = j; d < 512; d += 128) {
    float g = b2[d];
    for (int k = 0; k < 128; ++k) g += h[k] * w2[k * 512 + d];
    gate[b * 512 + d] = 1.0f / (1.0f + __expf(-g));
  }
}

// ---------------------------------------------------------------------------
__global__ __launch_bounds__(256)
void fuse_ln_kernel(const float* __restrict__ qp, const float* __restrict__ co,
                    const float* __restrict__ gate,
                    const float* __restrict__ lng, const float* __restrict__ lnb,
                    float* __restrict__ out)
{
  __shared__ float red[256];
  const int tok = blockIdx.x, b = tok >> 10, t = threadIdx.x;
  const float* xq = qp + (long)tok * 512;
  const float* xc = co + (long)tok * 512;
  const float* g  = gate + b * 512;

  float x0 = xq[t]       + xc[t]       * g[t];
  float x1 = xq[t + 256] + xc[t + 256] * g[t + 256];

  red[t] = x0 + x1; __syncthreads();
  for (int s = 128; s > 0; s >>= 1) { if (t < s) red[t] += red[t+s]; __syncthreads(); }
  const float mu = red[0] * (1.0f / 512.0f);
  __syncthreads();

  const float d0 = x0 - mu, d1 = x1 - mu;
  red[t] = d0 * d0 + d1 * d1; __syncthreads();
  for (int s = 128; s > 0; s >>= 1) { if (t < s) red[t] += red[t+s]; __syncthreads(); }
  const float inv = rsqrtf(red[0] * (1.0f / 512.0f) + LN_EPS);

  float* o = out + (long)tok * 512;
  o[t]       = d0 * inv * lng[t]       + lnb[t];
  o[t + 256] = d1 * inv * lng[t + 256] + lnb[t + 256];
}

// ---------------------------------------------------------------------------
extern "C" void kernel_launch(void* const* d_in, const int* in_sizes, int n_in,
                              void* d_out, int out_size, void* d_ws, size_t ws_size,
                              hipStream_t stream)
{
  const float* q      = (const float*)d_in[0];
  const float* k      = (const float*)d_in[1];
  const float* v      = (const float*)d_in[2];
  const float* Wq     = (const float*)d_in[3];
  const float* Wk     = (const float*)d_in[4];
  const float* Wv     = (const float*)d_in[5];
  const float* Wo     = (const float*)d_in[6];
  const float* conv_w = (const float*)d_in[7];
  const float* conv_b = (const float*)d_in[8];
  const float* se_w1  = (const float*)d_in[9];
  const float* se_b1  = (const float*)d_in[10];
  const float* se_w2  = (const float*)d_in[11];
  const float* se_b2  = (const float*)d_in[12];
  const float* ln_g   = (const float*)d_in[13];
  const float* ln_b   = (const float*)d_in[14];
  float* out = (float*)d_out;

  // Workspace layout (float units). Total ~91 MB with reuse.
  float* ws = (float*)d_ws;
  const long MND  = 8192L * 512;            // 4,194,304 f32 elems
  const long MNDh = MND / 2;                // same tensor as bf16, in f32 units
  float*  qp   = ws;                        // [8192,512] f32 (alive to end)
  __bf16* kpb  = (__bf16*)(ws + MND);               // [8192,512] bf16
  __bf16* vpb  = (__bf16*)(ws + MND + MNDh);        // [8192,512] bf16
  float*  sc   = ws + 2 * MND;                      // [8,1024,1024] f32 scores/probs
  float*  at   = ws + 2 * MND + 2 * MND;            // [8192,512] f32 attn out
  float*  oo   = sc;                                // o-proj reuses scores (probs dead)
  float*  co   = at;                                // conv out reuses attn (dead after o)
  long woff = 5 * MND;                              // 20,971,520
  __bf16* wqb = (__bf16*)(ws + woff);               // 512x512 bf16 each
  __bf16* wkb = (__bf16*)(ws + woff + 131072);
  __bf16* wvb = (__bf16*)(ws + woff + 2 * 131072);
  __bf16* wob = (__bf16*)(ws + woff + 3 * 131072);
  __bf16* cwb = (__bf16*)(ws + woff + 4 * 131072);  // 9*512*512 bf16
  float* pooled = ws + woff + 4 * 131072 + 1179648; // [8,512]
  float* gatep  = pooled + 4096;                    // [8,512]

  const float inv_sqrt_d = 0.044194173824159216f;   // 1/sqrt(512)

  // 0) cast weights to bf16
  cast_bf16_kernel<<<256, 256, 0, stream>>>(Wq, wqb, 262144);
  cast_bf16_kernel<<<256, 256, 0, stream>>>(Wk, wkb, 262144);
  cast_bf16_kernel<<<256, 256, 0, stream>>>(Wv, wvb, 262144);
  cast_bf16_kernel<<<256, 256, 0, stream>>>(Wo, wob, 262144);
  cast_bf16_kernel<<<2304, 256, 0, stream>>>(conv_w, cwb, 2359296);

  // 1) projections: [8192,512] @ [512,512]; k,v written directly as bf16
  dim3 gProj(4, 64, 1), blk(256);
  gemm_bf16_kernel<<<gProj, blk, 0, stream>>>(q, wqb, qp,  512, 512, 512, 0, 0, 0, 512, 1.0f, 0, 0);
  gemm_bf16_kernel<<<gProj, blk, 0, stream>>>(k, wkb, kpb, 512, 512, 512, 0, 0, 0, 512, 1.0f, 0, 1);
  gemm_bf16_kernel<<<gProj, blk, 0, stream>>>(v, wvb, vpb, 512, 512, 512, 0, 0, 0, 512, 1.0f, 0, 1);

  // 2) scores = qp @ kp^T / sqrt(D), batched over 8 (B is [N,K] bf16)
  dim3 gSc(8, 8, 8);
  gemm_bf16_kernel<<<gSc, blk, 0, stream>>>(qp, kpb, sc, 512, 512, 1024,
                                            1024L*512, 1024L*512, 1024L*1024,
                                            512, inv_sqrt_d, 1, 0);

  // 3) softmax rows (8*1024 rows of 1024)
  softmax_kernel<<<8192, 256, 0, stream>>>(sc);

  // 4) attn = probs @ vp, batched (B is [K,N] bf16)
  dim3 gAt(4, 8, 8);
  gemm_bf16_kernel<<<gAt, blk, 0, stream>>>(sc, vpb, at, 1024, 512, 512,
                                            1024L*1024, 1024L*512, 1024L*512,
                                            1024, 1.0f, 0, 0);

  // 5) o = attn @ Wo  (writes into scores region)
  gemm_bf16_kernel<<<gProj, blk, 0, stream>>>(at, wob, oo, 512, 512, 512, 0, 0, 0, 512, 1.0f, 0, 0);

  // 6) conv 3x3 SAME (implicit GEMM, bf16 weights) + bias (writes into attn region)
  dim3 gCv(4, 64, 1);
  conv3x3_kernel<<<gCv, blk, 0, stream>>>(oo, cwb, conv_b, co);

  // 7) SE pool, 8) SE MLP
  se_pool_kernel<<<8, 512, 0, stream>>>(co, pooled);
  se_mlp_kernel<<<8, 128, 0, stream>>>(pooled, se_w1, se_b1, se_w2, se_b2, gatep);

  // 9) residual + scale + layernorm
  fuse_ln_kernel<<<8192, 256, 0, stream>>>(qp, co, gatep, ln_g, ln_b, out);
}
